// AttentionAll_11699490914515
// MI455X (gfx1250) — matrix operand order repaired and checked by reading, compiled, not run on hardware
//
#include <hip/hip_runtime.h>
#include <hip/hip_bf16.h>

// ---------------------------------------------------------------------------
// ViT attention block with policy-masked softmax, bf16 WMMA pipeline (gfx1250)
//   B=16, N=577 (padded 592 rows / 608 keys), DIM=768, H=12, Dh=64
//   GEMM waves compute 16x64 tiles: 1 A-frag -> 4 independent WMMA chains,
//   with explicit double-buffering of fragments to overlap loads with WMMAs.
// ---------------------------------------------------------------------------

typedef __bf16 bf16_t;
typedef __attribute__((ext_vector_type(16))) __bf16 v16bf;
typedef __attribute__((ext_vector_type(8)))  __bf16 v8bf;
typedef __attribute__((ext_vector_type(8)))  float  v8f;

#define NPAD   608                 // key padding (19 * 32)
#define MPAD   592                 // query/row padding (37 * 16)
#define NTOK   577
#define NDIM   768
#define NHEAD  12
#define HDIM   64
#define SCALE  0.125f              // 64^-0.5
#define EPSV   1e-6f
#define EPSN   (1e-6f / 577.0f)

__device__ __forceinline__ v8f wmma_bf16(v16bf a, v16bf b, v8f c) {
    return __builtin_amdgcn_wmma_f32_16x16x32_bf16(
        /*neg_a=*/false, a, /*neg_b=*/false, b,
        /*c_mod=*/(short)0, c, /*reuse_a=*/false, /*reuse_b=*/false);
}

// A fragment (16x32, bf16, row-major, leading dim ld). tile -> &A[0][k0].
// Lane L: row m = L%16, half = L/16. VGPR0-7 <-> K = half*16+0..15
// (two contiguous 16B loads).
__device__ __forceinline__ v16bf load_a_rowmajor(const bf16_t* tile, int ld) {
    int lane = threadIdx.x & 31;
    int m = lane & 15, hf = lane >> 4;
    const bf16_t* p = tile + m * ld + hf * 16;
    v8bf lo = *reinterpret_cast<const v8bf*>(p);
    v8bf hi = *reinterpret_cast<const v8bf*>(p + 8);
    v16bf a;
#pragma unroll
    for (int i = 0; i < 8; ++i) { a[i] = lo[i]; a[i + 8] = hi[i]; }
    return a;
}

// B fragment (32x16, bf16) from K-contiguous storage: bt row = output column n,
// contiguous contraction. Lane L: n = L%16, half = L/16, 16 contiguous K at
// bt[n*ld + half*16]  (one 32B load).
__device__ __forceinline__ v16bf load_b_kcontig(const bf16_t* bt, int ld) {
    int lane = threadIdx.x & 31;
    int n = lane & 15, hf = lane >> 4;
    return *reinterpret_cast<const v16bf*>(bt + n * ld + hf * 16);
}

// ---------------------------------------------------------------------------
// 1) fp32 -> bf16 conversions: x (row-padded, pad rows zeroed), Wqkv^T, Wproj^T
// ---------------------------------------------------------------------------
__global__ void k_convert(const float* __restrict__ x,
                          const float* __restrict__ Wqkv,
                          const float* __restrict__ Wproj,
                          bf16_t* __restrict__ xbf,
                          bf16_t* __restrict__ WqkvT,
                          bf16_t* __restrict__ WprojT) {
    size_t i = (size_t)blockIdx.x * blockDim.x + threadIdx.x;
    const size_t n1 = (size_t)3 * NDIM * NDIM;            // WqkvT: [2304][768]
    const size_t n2 = (size_t)NDIM * NDIM;                // WprojT: [768][768]
    const size_t n3 = (size_t)16 * MPAD * NDIM;           // xbf: [16][592][768]
    if (i < n1) {
        size_t n = i / NDIM, k = i % NDIM;
        WqkvT[i] = (bf16_t)Wqkv[k * (3 * NDIM) + n];
    } else if (i < n1 + n2) {
        size_t j = i - n1;
        size_t n = j / NDIM, k = j % NDIM;
        WprojT[j] = (bf16_t)Wproj[k * NDIM + n];
    } else if (i < n1 + n2 + n3) {
        size_t j = i - n1 - n2;
        size_t b = j / ((size_t)MPAD * NDIM);
        size_t r = (j / NDIM) % MPAD;
        size_t c = j % NDIM;
        xbf[j] = (r < NTOK) ? (bf16_t)x[(b * NTOK + r) * NDIM + c] : (bf16_t)0.f;
    }
}

// Double-buffered 16x64 GEMM tile body over K = NDIM (24 k-steps of 32).
// A row-major [16 x NDIM] at Ap (ld NDIM); B K-contiguous rows at Btp.
__device__ __forceinline__ void gemm_16x64(const bf16_t* Ap, const bf16_t* Btp,
                                           v8f& a0, v8f& a1, v8f& a2, v8f& a3) {
    v16bf af = load_a_rowmajor(Ap, NDIM);
    v16bf b0 = load_b_kcontig(Btp + 0 * 16 * NDIM, NDIM);
    v16bf b1 = load_b_kcontig(Btp + 1 * 16 * NDIM, NDIM);
    v16bf b2 = load_b_kcontig(Btp + 2 * 16 * NDIM, NDIM);
    v16bf b3 = load_b_kcontig(Btp + 3 * 16 * NDIM, NDIM);
    for (int kk = 0; kk < NDIM - 32; kk += 32) {
        int kn = kk + 32;
        // issue next k-step's loads first, then compute on current fragments
        v16bf af_n = load_a_rowmajor(Ap + kn, NDIM);
        v16bf b0_n = load_b_kcontig(Btp + 0 * 16 * NDIM + kn, NDIM);
        v16bf b1_n = load_b_kcontig(Btp + 1 * 16 * NDIM + kn, NDIM);
        v16bf b2_n = load_b_kcontig(Btp + 2 * 16 * NDIM + kn, NDIM);
        v16bf b3_n = load_b_kcontig(Btp + 3 * 16 * NDIM + kn, NDIM);
        a0 = wmma_bf16(af, b0, a0);
        a1 = wmma_bf16(af, b1, a1);
        a2 = wmma_bf16(af, b2, a2);
        a3 = wmma_bf16(af, b3, a3);
        af = af_n; b0 = b0_n; b1 = b1_n; b2 = b2_n; b3 = b3_n;
    }
    a0 = wmma_bf16(af, b0, a0);
    a1 = wmma_bf16(af, b1, a1);
    a2 = wmma_bf16(af, b2, a2);
    a3 = wmma_bf16(af, b3, a3);
}

// ---------------------------------------------------------------------------
// 2) QKV projection: [B*592, 768] x [768, 2304], one wave per 16x64 tile
//    (64-col span == one head's worth of a q/k/v section -> uniform scatter).
//    Scatter into Q [B,H,592,64], K [B,H,592,64], Vt [B,H,64,608].
// ---------------------------------------------------------------------------
__global__ void __launch_bounds__(128)
k_qkv(const bf16_t* __restrict__ xbf, const bf16_t* __restrict__ WqkvT,
      const float* __restrict__ bqkv,
      bf16_t* __restrict__ Q, bf16_t* __restrict__ Kh, bf16_t* __restrict__ Vt) {
    int wave = threadIdx.x >> 5;
    int lane = threadIdx.x & 31;
    int ct = blockIdx.x * 4 + wave;            // 0..35, each covers 64 columns
    int mt = blockIdx.y;                       // 0..36 row tiles
    int b  = blockIdx.z;
    int c0 = ct * 64;
    const bf16_t* A  = xbf   + ((size_t)b * MPAD + mt * 16) * NDIM;
    const bf16_t* Bt = WqkvT + (size_t)c0 * NDIM;

    v8f a0 = {}, a1 = {}, a2 = {}, a3 = {};
    gemm_16x64(A, Bt, a0, a1, a2, a3);

    int n = lane & 15, hf = lane >> 4;
    int sec = c0 / NDIM;                       // 0=q 1=k 2=v (uniform in wave)
    int h   = (c0 % NDIM) >> 6;                // head (uniform in wave)
    size_t bh = (size_t)b * NHEAD + h;
    v8f acc[4] = {a0, a1, a2, a3};
#pragma unroll
    for (int dc = 0; dc < 4; ++dc) {
        int d = dc * 16 + n;                   // dim within head, 0..63
        float bias = bqkv[c0 + d];
#pragma unroll
        for (int v = 0; v < 8; ++v) {
            int row = mt * 16 + v + hf * 8;
            bf16_t o = (bf16_t)(acc[dc][v] + bias);
            if (sec == 0)      Q [(bh * MPAD + row) * HDIM + d] = o;
            else if (sec == 1) Kh[(bh * MPAD + row) * HDIM + d] = o;
            else               Vt[(bh * HDIM + d) * NPAD + row] = o;
        }
    }
}

// ---------------------------------------------------------------------------
// 3) Attention: one wave per (b, h, 16-query tile).
//    S = Q K^T * scale (fp32 in LDS) -> policy-masked softmax -> P (bf16 LDS)
//    -> O = P V (WMMA, 4 accumulator chains) -> /denominator -> AO bf16
// ---------------------------------------------------------------------------
__global__ void __launch_bounds__(32)
k_attn(const bf16_t* __restrict__ Q, const bf16_t* __restrict__ Kh,
       const bf16_t* __restrict__ Vt, const float* __restrict__ policy,
       bf16_t* __restrict__ AO) {
    extern __shared__ char smem[];
    float*  S   = (float*)smem;                               // [16][608] fp32
    bf16_t* P   = (bf16_t*)(smem + 16 * NPAD * 4);            // [16][608] bf16
    float*  pol = (float*)(smem + 16 * NPAD * 4 + 16 * NPAD * 2); // [608]
    float*  red = pol + NPAD;                                 // [16]

    int lane = threadIdx.x;
    int mt = blockIdx.x, h = blockIdx.y, b = blockIdx.z;
    size_t bh = (size_t)b * NHEAD + h;
    int n = lane & 15, hf = lane >> 4;

    // cache the policy row (pad -> 0)
    for (int c = lane; c < NPAD; c += 32)
        pol[c] = (c < NTOK) ? policy[(size_t)b * NTOK + c] : 0.f;

    // Q fragments for this row tile (K = 0..31 and 32..63)
    const bf16_t* Qb = Q + (bh * MPAD + (size_t)mt * 16) * HDIM;
    v16bf aq0 = load_a_rowmajor(Qb, HDIM);
    v16bf aq1 = load_a_rowmajor(Qb + 32, HDIM);

    // S = Q K^T * scale, with next K-tile prefetch
    const bf16_t* Kb0 = Kh + bh * MPAD * HDIM;
    v16bf kb0 = load_b_kcontig(Kb0, HDIM);
    v16bf kb1 = load_b_kcontig(Kb0 + 32, HDIM);
    for (int j = 0; j < 37; ++j) {
        v16bf kn0 = kb0, kn1 = kb1;
        if (j + 1 < 37) {
            const bf16_t* Kb = Kb0 + (size_t)(j + 1) * 16 * HDIM;
            kn0 = load_b_kcontig(Kb, HDIM);
            kn1 = load_b_kcontig(Kb + 32, HDIM);
        }
        v8f s = {};
        s = wmma_bf16(aq0, kb0, s);
        s = wmma_bf16(aq1, kb1, s);
#pragma unroll
        for (int v = 0; v < 8; ++v)
            S[(v + hf * 8) * NPAD + j * 16 + n] = s[v] * SCALE;
        kb0 = kn0; kb1 = kn1;
    }
    __syncthreads();

    // row-wise masked softmax: lanes r and r+16 split the columns of row r
    {
        int r = lane & 15;
        int q = mt * 16 + r;                  // global query index (eye term)
        float mx = -3.0e38f;
        for (int c = hf; c < NTOK; c += 2)
            mx = fmaxf(mx, S[r * NPAD + c]);
        mx = fmaxf(mx, __shfl_xor(mx, 16, 32));
        float sum = 0.f;
        for (int c = hf; c < NPAD; c += 2) {
            float pv = 0.f;
            if (c < NTOK) {
                float ap = pol[c];
                if (c == q) ap = 1.f;          // ap + (1-ap)*eye
                float e = __expf(S[r * NPAD + c] - mx) * ap;
                sum += e;
                pv = e + EPSN;                 // numerator: e + EPS/N
            }
            P[r * NPAD + c] = (bf16_t)pv;      // pad columns forced to 0
        }
        sum += __shfl_xor(sum, 16, 32);
        if (hf == 0) red[r] = sum + EPSV;      // denominator
    }
    __syncthreads();

    // O = P @ V  (A from LDS, B from K-contiguous Vt), Dh=64 -> 4 C tiles,
    // with next V-fragment prefetch (global); A frag comes from fast LDS.
    v8f o0 = {}, o1 = {}, o2 = {}, o3 = {};
    const bf16_t* Vb = Vt + (bh * HDIM) * NPAD;
    v16bf vb0 = load_b_kcontig(Vb + 0 * 16 * NPAD, NPAD);
    v16bf vb1 = load_b_kcontig(Vb + 1 * 16 * NPAD, NPAD);
    v16bf vb2 = load_b_kcontig(Vb + 2 * 16 * NPAD, NPAD);
    v16bf vb3 = load_b_kcontig(Vb + 3 * 16 * NPAD, NPAD);
    for (int kk = 0; kk < NPAD; kk += 32) {
        v16bf n0 = vb0, n1 = vb1, n2 = vb2, n3 = vb3;
        int kn = kk + 32;
        if (kn < NPAD) {
            n0 = load_b_kcontig(Vb + 0 * 16 * NPAD + kn, NPAD);
            n1 = load_b_kcontig(Vb + 1 * 16 * NPAD + kn, NPAD);
            n2 = load_b_kcontig(Vb + 2 * 16 * NPAD + kn, NPAD);
            n3 = load_b_kcontig(Vb + 3 * 16 * NPAD + kn, NPAD);
        }
        v16bf a = load_a_rowmajor(P + kk, NPAD);
        o0 = wmma_bf16(a, vb0, o0);
        o1 = wmma_bf16(a, vb1, o1);
        o2 = wmma_bf16(a, vb2, o2);
        o3 = wmma_bf16(a, vb3, o3);
        vb0 = n0; vb1 = n1; vb2 = n2; vb3 = n3;
    }
#pragma unroll
    for (int v = 0; v < 8; ++v) {
        int r = v + hf * 8;
        float inv = 1.f / red[r];
        int row = mt * 16 + r;
        bf16_t* dst = AO + ((size_t)b * MPAD + row) * NDIM + h * HDIM + n;
        dst[0]  = (bf16_t)(o0[v] * inv);
        dst[16] = (bf16_t)(o1[v] * inv);
        dst[32] = (bf16_t)(o2[v] * inv);
        dst[48] = (bf16_t)(o3[v] * inv);
    }
}

// ---------------------------------------------------------------------------
// 4) Output projection: [B*592, 768] x [768, 768] -> fp32 out (valid rows only)
//    One wave per 16x64 tile, 4 accumulator chains, double-buffered.
// ---------------------------------------------------------------------------
__global__ void __launch_bounds__(128)
k_proj(const bf16_t* __restrict__ AO, const bf16_t* __restrict__ WprojT,
       const float* __restrict__ bproj, float* __restrict__ out) {
    int wave = threadIdx.x >> 5;
    int lane = threadIdx.x & 31;
    int ct = blockIdx.x * 4 + wave;           // 0..11, each covers 64 columns
    int mt = blockIdx.y;                      // 0..36
    int b  = blockIdx.z;
    int c0 = ct * 64;
    const bf16_t* A  = AO     + ((size_t)b * MPAD + mt * 16) * NDIM;
    const bf16_t* Bt = WprojT + (size_t)c0 * NDIM;

    v8f a0 = {}, a1 = {}, a2 = {}, a3 = {};
    gemm_16x64(A, Bt, a0, a1, a2, a3);

    int n = lane & 15, hf = lane >> 4;
    v8f acc[4] = {a0, a1, a2, a3};
#pragma unroll
    for (int dc = 0; dc < 4; ++dc) {
        int col = c0 + dc * 16 + n;
        float bias = bproj[col];
#pragma unroll
        for (int v = 0; v < 8; ++v) {
            int row = mt * 16 + v + hf * 8;
            if (row < NTOK)
                out[((size_t)b * NTOK + row) * NDIM + col] = acc[dc][v] + bias;
        }
    }
}

// ---------------------------------------------------------------------------
extern "C" void kernel_launch(void* const* d_in, const int* in_sizes, int n_in,
                              void* d_out, int out_size, void* d_ws, size_t ws_size,
                              hipStream_t stream) {
    const float* x      = (const float*)d_in[0];   // [16,577,768]
    const float* policy = (const float*)d_in[1];   // [16,577,1]
    const float* Wqkv   = (const float*)d_in[2];   // [768,2304]
    const float* bqkv   = (const float*)d_in[3];   // [2304]
    const float* Wproj  = (const float*)d_in[4];   // [768,768]
    const float* bproj  = (const float*)d_in[5];   // [768]
    float* out = (float*)d_out;                    // [16,577,768]

    char* ws = (char*)d_ws;
    // workspace layout (all offsets 32B aligned)
    bf16_t* xbf    = (bf16_t*)(ws + 0);            // 16*592*768*2  = 14,548,992
    bf16_t* WqkvT  = (bf16_t*)(ws + 14548992);     // 2304*768*2    =  3,538,944
    bf16_t* WprojT = (bf16_t*)(ws + 18087936);     // 768*768*2     =  1,179,648
    bf16_t* Q      = (bf16_t*)(ws + 19267584);     // 16*12*592*64*2 = 14,548,992
    bf16_t* Kh     = (bf16_t*)(ws + 33816576);     // same
    bf16_t* Vt     = (bf16_t*)(ws + 48365568);     // 16*12*64*608*2 = 14,942,208
    bf16_t* AO     = (bf16_t*)(ws + 63307776);     // 16*592*768*2  = 14,548,992

    size_t total = (size_t)3 * NDIM * NDIM + (size_t)NDIM * NDIM
                 + (size_t)16 * MPAD * NDIM;
    k_convert<<<(unsigned)((total + 255) / 256), 256, 0, stream>>>(
        x, Wqkv, Wproj, xbf, WqkvT, WprojT);

    k_qkv<<<dim3(9, 37, 16), 128, 0, stream>>>(xbf, WqkvT, bqkv, Q, Kh, Vt);

    size_t shbytes = (size_t)16 * NPAD * 4 + (size_t)16 * NPAD * 2
                   + (size_t)NPAD * 4 + 64;        // 60,864 B
    k_attn<<<dim3(37, NHEAD, 16), 32, shbytes, stream>>>(Q, Kh, Vt, policy, AO);

    k_proj<<<dim3(3, 37, 16), 128, 0, stream>>>(AO, WprojT, bproj, out);
}
